// CrossAttentionExtractor_18339510354835
// MI455X (gfx1250) — matrix operand order repaired and checked by reading, not compiled, not run on hardware
//
#include <hip/hip_runtime.h>
#include <hip/hip_bf16.h>

#define B_   8
#define T_   2
#define Q_   100
#define C_   256
#define H_   64
#define W_   64
#define NH_  8
#define DH_  32
#define DF_  2048
#define NI_  10
#define BT_  16
#define HW_  4096
#define G_   32
#define MR_  (BT_ * Q_)   /* 1600 rows in decoder stage */
#define KCH_ 32           /* GEMM k-chunk staged in LDS */

typedef __attribute__((ext_vector_type(16))) _Float16 v16h;
typedef __attribute__((ext_vector_type(8)))  _Float16 v8h;
typedef __attribute__((ext_vector_type(8)))  float    v8f;
typedef __attribute__((ext_vector_type(4)))  int      v4i;

// ---------------------------------------------------------------------------
// CDNA5 async global->LDS staging (guarded; falls back to plain LDS stores)
// Builtin signature (probe-confirmed): (v4i as(1)* gsrc, v4i as(3)* ldst,
// imm offset, imm cpol).
// ---------------------------------------------------------------------------
#if defined(__gfx1250__) && __has_builtin(__builtin_amdgcn_global_load_async_to_lds_b128)
#define ASYNC_LDS 1
#endif

__device__ __forceinline__ void stage16(const _Float16* g, _Float16* l) {
#ifdef ASYNC_LDS
  __builtin_amdgcn_global_load_async_to_lds_b128(
      (__attribute__((address_space(1))) v4i*)g,
      (__attribute__((address_space(3))) v4i*)l, 0, 0);
#else
  *(v8h*)l = *(const v8h*)g;
#endif
}

template <int N>
__device__ __forceinline__ void stage_wait() {
#ifdef ASYNC_LDS
#if __has_builtin(__builtin_amdgcn_s_wait_asynccnt)
  __builtin_amdgcn_s_wait_asynccnt(N);
#else
  asm volatile("s_wait_asynccnt %0" ::"i"(N));
#endif
#endif
}

// ---------------------------------------------------------------------------
// Positional sine embedding: pos[hw][c]
// ---------------------------------------------------------------------------
__global__ void pos_embed_kernel(float* __restrict__ pos) {
  int idx = blockIdx.x * blockDim.x + threadIdx.x;
  if (idx >= HW_ * C_) return;
  int c = idx % C_, hw = idx / C_;
  int h = hw / W_, w = hw % W_;
  const float twopi = 6.283185307179586f;
  float coord = (c < C_ / 2) ? ((float)(h + 1) / (H_ + 1e-6f)) * twopi
                             : ((float)(w + 1) / (W_ + 1e-6f)) * twopi;
  int cc = (c < C_ / 2) ? c : c - C_ / 2;
  int k = cc >> 1;
  float dimt = __powf(10000.0f, (2.0f * (float)k) / (float)(C_ / 2));
  float p = coord / dimt;
  pos[idx] = (cc & 1) ? __cosf(p) : __sinf(p);
}

// ---------------------------------------------------------------------------
// f32 -> f16 convert
// ---------------------------------------------------------------------------
__global__ void cvt_f16_kernel(const float* __restrict__ src, _Float16* __restrict__ dst, int n) {
  int i = blockIdx.x * blockDim.x + threadIdx.x;
  if (i < n) dst[i] = (_Float16)src[i];
}

// ---------------------------------------------------------------------------
// features [b][C][HW] f32 -> featT16 [b][HW][C] f16 (LDS tiled transpose)
// ---------------------------------------------------------------------------
__global__ __launch_bounds__(256)
void transpose_f16_kernel(const float* __restrict__ in, _Float16* __restrict__ out) {
  __shared__ float tile[32][33];
  const long ib = (long)blockIdx.z * C_ * HW_;
  const int hw0 = blockIdx.x * 32, c0 = blockIdx.y * 32;
  const int tx = threadIdx.x & 31, ty = threadIdx.x >> 5;
  #pragma unroll
  for (int r = ty; r < 32; r += 8)
    tile[r][tx] = in[ib + (long)(c0 + r) * HW_ + hw0 + tx];
  __syncthreads();
  #pragma unroll
  for (int r = ty; r < 32; r += 8)
    out[ib + (long)(hw0 + r) * C_ + c0 + tx] = (_Float16)tile[tx][r];
}

// ---------------------------------------------------------------------------
// Generic WMMA GEMM:  D[m][n] = sum_k A[m][k]*W[n][k] + bias[n]
// A f16 [M][K] staged through LDS with double-buffered async copies;
// W f16 [N][K] streamed from global (L2-resident weights).
// 64x64 block tile, 32x32 per wave (4 waves). Optional f32 / f16 (optionally
// transposed [N][M]) outputs, optional relu.
// ---------------------------------------------------------------------------
__global__ __launch_bounds__(128)
void gemm_wmma_kernel(const _Float16* __restrict__ A, const _Float16* __restrict__ W,
                      const float* __restrict__ bias,
                      float* __restrict__ out32, _Float16* __restrict__ out16,
                      int M, int N, int K,
                      long aBatchStride, long oBatchStride,
                      int relu, int out16T) {
  __shared__ _Float16 sA[2][64][KCH_];  // 2 x 4 KB double buffer
  const int wave = threadIdx.x >> 5;
  const int lane = threadIdx.x & 31;
  const int lh = lane & 15;
  const int hi = lane >> 4;
  const long ab = (long)blockIdx.z * aBatchStride;
  const long ob = (long)blockIdx.z * oBatchStride;

  const int m0 = blockIdx.x * 64;
  const int mw = (wave & 1) * 32;                 // wave M offset within tile
  const int n0 = blockIdx.y * 64 + (wave >> 1) * 32;

  // staging split: thread covers rows tr and tr+32, 8 halves at column seg
  const int tr = threadIdx.x >> 2;        // 0..31
  const int seg = (threadIdx.x & 3) * 8;  // 0,8,16,24
  const _Float16* Abase = A + ab;

  // prologue: stage chunk 0
  stage16(Abase + (long)(m0 + tr) * K + seg, &sA[0][tr][seg]);
  stage16(Abase + (long)(m0 + 32 + tr) * K + seg, &sA[0][32 + tr][seg]);

  const int nch = K / KCH_;
  v8f acc[2][2] = {};
  for (int ch = 0; ch < nch; ++ch) {
    const int buf = ch & 1;
    if (ch + 1 < nch) {
      const int kn = (ch + 1) * KCH_;
      stage16(Abase + (long)(m0 + tr) * K + kn + seg, &sA[buf ^ 1][tr][seg]);
      stage16(Abase + (long)(m0 + 32 + tr) * K + kn + seg, &sA[buf ^ 1][32 + tr][seg]);
      stage_wait<2>();  // chunk `ch` (2 older async ops) complete
    } else {
      stage_wait<0>();
    }
    __syncthreads();

    const int kk = ch * KCH_;
    v16h a[2], bm[2];
    #pragma unroll
    for (int i = 0; i < 2; ++i) {
      const _Float16* ap = &sA[buf][mw + i * 16 + lh][hi ? 16 : 0];
      v8h a0 = *(const v8h*)(ap);
      v8h a1 = *(const v8h*)(ap + 8);
      #pragma unroll
      for (int e = 0; e < 8; ++e) { a[i][e] = a0[e]; a[i][8 + e] = a1[e]; }
      const _Float16* wp = W + (long)(n0 + i * 16 + lh) * K + kk + (hi ? 16 : 0);
      bm[i] = *(const v16h*)(wp);
    }
    #pragma unroll
    for (int i = 0; i < 2; ++i)
      #pragma unroll
      for (int j = 0; j < 2; ++j)
        acc[i][j] = __builtin_amdgcn_wmma_f32_16x16x32_f16(
            false, a[i], false, bm[j], (short)0, acc[i][j], false, false);
    __syncthreads();  // reads of buf done before it is restaged
  }

  #pragma unroll
  for (int i = 0; i < 2; ++i) {
    #pragma unroll
    for (int j = 0; j < 2; ++j) {
      const int col = n0 + j * 16 + lh;
      const float bv = bias ? bias[col] : 0.0f;
      #pragma unroll
      for (int r = 0; r < 8; ++r) {
        const int row = m0 + mw + i * 16 + (hi ? 8 : 0) + r;
        float v = acc[i][j][r] + bv;
        if (relu) v = v > 0.0f ? v : 0.0f;
        if (out32) out32[ob + (long)row * N + col] = v;
        if (out16) {
          if (out16T) out16[ob + (long)col * M + row] = (_Float16)v;
          else        out16[ob + (long)row * N + col] = (_Float16)v;
        }
      }
    }
  }
}

// ---------------------------------------------------------------------------
// GroupNorm stats -> per-(b,c) scale/shift
// ---------------------------------------------------------------------------
__global__ __launch_bounds__(256)
void gn_stats_kernel(const _Float16* __restrict__ f, const float* __restrict__ gamma,
                     const float* __restrict__ beta, float* __restrict__ scale,
                     float* __restrict__ shift) {
  __shared__ float ss[256], ss2[256];
  const int b = blockIdx.x / G_, g = blockIdx.x % G_;
  const int cg = C_ / G_;  // 8
  float s = 0.f, s2 = 0.f;
  const long base = (long)b * HW_ * C_ + g * cg;
  for (int hw = threadIdx.x; hw < HW_; hw += 256) {
    const _Float16* p = f + base + (long)hw * C_;
    #pragma unroll
    for (int j = 0; j < 8; ++j) { float v = (float)p[j]; s += v; s2 += v * v; }
  }
  ss[threadIdx.x] = s; ss2[threadIdx.x] = s2;
  __syncthreads();
  for (int st = 128; st > 0; st >>= 1) {
    if (threadIdx.x < st) { ss[threadIdx.x] += ss[threadIdx.x + st]; ss2[threadIdx.x] += ss2[threadIdx.x + st]; }
    __syncthreads();
  }
  if (threadIdx.x < 8) {
    const float n = (float)(HW_ * cg);
    const float mu = ss[0] / n;
    const float var = ss2[0] / n - mu * mu;
    const float rstd = rsqrtf(var + 1e-5f);
    const int c = g * cg + threadIdx.x;
    const float sc = gamma[c] * rstd;
    scale[b * C_ + c] = sc;
    shift[b * C_ + c] = beta[c] - mu * sc;
  }
}

// mem16 = GN(f);  kin16 = GN(f) + pos
__global__ void mem_build_kernel(const _Float16* __restrict__ f, const float* __restrict__ scale,
                                 const float* __restrict__ shift, const float* __restrict__ pos,
                                 _Float16* __restrict__ mem16, _Float16* __restrict__ kin16) {
  long idx = (long)blockIdx.x * blockDim.x + threadIdx.x;
  if (idx >= (long)BT_ * HW_ * C_) return;
  int c = (int)(idx % C_);
  long bhw = idx / C_;
  int hw = (int)(bhw % HW_);
  int b = (int)(bhw / HW_);
  float m = (float)f[idx] * scale[b * C_ + c] + shift[b * C_ + c];
  mem16[idx] = (_Float16)m;
  kin16[idx] = (_Float16)(m + pos[hw * C_ + c]);
}

// ---------------------------------------------------------------------------
// Mask pipeline
// ---------------------------------------------------------------------------
__global__ void allmask_kernel(const float* __restrict__ tm, float* __restrict__ allm) {
  int idx = blockIdx.x * blockDim.x + threadIdx.x;
  if (idx >= BT_ * HW_) return;
  int b = idx / HW_, hw = idx % HW_;
  const float* p = tm + (long)b * NI_ * HW_ + hw;
  float m = -3.0e38f;
  for (int i = 0; i < NI_; ++i) m = fmaxf(m, p[(long)i * HW_]);
  allm[idx] = m;
}

__global__ void maskb_kernel(const float* __restrict__ tm, const float* __restrict__ sn,
                             const int* __restrict__ mi, const float* __restrict__ allm,
                             unsigned char* __restrict__ mb) {
  long idx = (long)blockIdx.x * blockDim.x + threadIdx.x;
  if (idx >= (long)BT_ * Q_ * HW_) return;
  int hw = (int)(idx % HW_);
  long bq = idx / HW_;
  int q = (int)(bq % Q_);
  int b = (int)(bq / Q_);
  int i = mi[b * Q_ + q];
  float v;
  if (i < NI_) v = tm[((long)b * NI_ + i) * HW_ + hw];
  else         v = sn[((long)b * Q_ + i) * HW_ + hw] * ((allm[b * HW_ + hw] == 0.0f) ? 1.0f : 0.0f);
  mb[idx] = (v < 0.0f) ? 1 : 0;  // sigmoid(v)<0.5 <=> v<0
}

__global__ __launch_bounds__(256)
void rowfull_kernel(const unsigned char* __restrict__ mb, unsigned char* __restrict__ rf) {
  __shared__ int sh[256];
  const long base = (long)blockIdx.x * HW_;
  int all1 = 1;
  for (int k = threadIdx.x; k < HW_; k += 256) all1 &= (int)mb[base + k];
  sh[threadIdx.x] = all1;
  __syncthreads();
  for (int st = 128; st > 0; st >>= 1) {
    if (threadIdx.x < st) sh[threadIdx.x] &= sh[threadIdx.x + st];
    __syncthreads();
  }
  if (threadIdx.x == 0) rf[blockIdx.x] = (unsigned char)sh[0];
}

// ---------------------------------------------------------------------------
// qh = (query_feat + query_embed) @ wq^T + bq   (f32 dot, stored f16)
// ---------------------------------------------------------------------------
__global__ __launch_bounds__(256)
void qh_kernel(const float* __restrict__ qf, const float* __restrict__ qe,
               const float* __restrict__ wq, const float* __restrict__ bq,
               _Float16* __restrict__ qh16) {
  int q = blockIdx.x, c = threadIdx.x;
  float acc = bq[c];
  for (int k = 0; k < C_; ++k)
    acc += (qf[q * C_ + k] + qe[q * C_ + k]) * wq[c * C_ + k];
  qh16[q * C_ + c] = (_Float16)acc;
}

// ---------------------------------------------------------------------------
// Fused masked attention: one workgroup per (q-tile, head, batch).
// Dynamic LDS: 16 x 4096 f32 score buffer (256 KB; WGP has 320 KB).
// ---------------------------------------------------------------------------
__global__ __launch_bounds__(128)
void attention_kernel(const _Float16* __restrict__ qh16, const _Float16* __restrict__ kh16,
                      const _Float16* __restrict__ vh16T, const unsigned char* __restrict__ mb,
                      const unsigned char* __restrict__ rf, _Float16* __restrict__ ctx16) {
  extern __shared__ float sscore[];       // [16][HW_]
  __shared__ float sred[16][9];
  __shared__ float cpart[4][16][32];
  const int wave = threadIdx.x >> 5, lane = threadIdx.x & 31;
  const int lh = lane & 15, hi = lane >> 4;
  const int q0 = blockIdx.x * 16;
  const int h = blockIdx.y;
  const int b = blockIdx.z;
  const float sinv = 0.17677669529663687f;  // 1/sqrt(32)

  // A operand: 16 query rows x 32 (=dh) — constant over the whole k loop.
  v16h aq;
  {
    int row = q0 + lh; if (row >= Q_) row = Q_ - 1;
    const _Float16* ap = qh16 + (long)row * C_ + h * DH_ + (hi ? 16 : 0);
    v8h a0 = *(const v8h*)ap;
    v8h a1 = *(const v8h*)(ap + 8);
    #pragma unroll
    for (int e = 0; e < 8; ++e) { aq[e] = a0[e]; aq[8 + e] = a1[e]; }
  }

  // Phase 1: scores into LDS (each wave covers HW/4 columns).
  const long khb = (long)b * HW_ * C_;
  const int kbeg = wave * (HW_ / 4), kend = kbeg + HW_ / 4;
  for (int kb = kbeg; kb < kend; kb += 16) {
    const _Float16* kp = kh16 + khb + (long)(kb + lh) * C_ + h * DH_ + (hi ? 16 : 0);
    __builtin_prefetch(kp + 16 * C_, 0, 1);  // next K tile -> global_prefetch
    v16h bv = *(const v16h*)kp;
    v8f s = {};
    s = __builtin_amdgcn_wmma_f32_16x16x32_f16(false, aq, false, bv, (short)0, s, false, false);
    const int col = kb + lh;
    #pragma unroll
    for (int r = 0; r < 8; ++r) {
      const int rt = (hi ? 8 : 0) + r;
      int qr = q0 + rt; if (qr >= Q_) qr = Q_ - 1;
      float v = s[r] * sinv;
      if (mb[((long)b * Q_ + qr) * HW_ + col] && !rf[b * Q_ + qr]) v = -1e30f;
      sscore[rt * HW_ + col] = v;
    }
  }
  __syncthreads();

  // Phase 2: row softmax (8 threads per row).
  {
    const int row = threadIdx.x >> 3, seg = threadIdx.x & 7;
    float* sr = sscore + row * HW_;
    const int k0 = seg * (HW_ / 8), k1 = k0 + HW_ / 8;
    float mx = -3.0e38f;
    for (int k = k0; k < k1; ++k) mx = fmaxf(mx, sr[k]);
    sred[row][seg] = mx;
    __syncthreads();
    if (seg == 0) { float m = -3.0e38f; for (int j = 0; j < 8; ++j) m = fmaxf(m, sred[row][j]); sred[row][8] = m; }
    __syncthreads();
    const float rm = sred[row][8];
    float sum = 0.f;
    for (int k = k0; k < k1; ++k) { float e = __expf(sr[k] - rm); sr[k] = e; sum += e; }
    sred[row][seg] = sum;
    __syncthreads();
    if (seg == 0) { float t = 0.f; for (int j = 0; j < 8; ++j) t += sred[row][j]; sred[row][8] = 1.0f / t; }
    __syncthreads();
    const float inv = sred[row][8];
    for (int k = k0; k < k1; ++k) sr[k] *= inv;
  }
  __syncthreads();

  // Phase 3: ctx = P @ V  (A from LDS probabilities, B from transposed V).
  const _Float16* vb = vh16T + (long)b * C_ * HW_ + (long)(h * DH_) * HW_;
  v8f acc0 = {}, acc1 = {};
  for (int kb = kbeg; kb < kend; kb += 32) {
    v16h pa;
    {
      const float* sp = sscore + lh * HW_ + kb + (hi ? 16 : 0);
      #pragma unroll
      for (int e = 0; e < 8; ++e) { pa[e] = (_Float16)sp[e]; pa[8 + e] = (_Float16)sp[e + 8]; }
    }
    const _Float16* vp0 = vb + (long)lh * HW_ + kb + (hi ? 16 : 0);
    v16h bv0 = *(const v16h*)vp0;
    acc0 = __builtin_amdgcn_wmma_f32_16x16x32_f16(false, pa, false, bv0, (short)0, acc0, false, false);
    const _Float16* vp1 = vb + (long)(16 + lh) * HW_ + kb + (hi ? 16 : 0);
    v16h bv1 = *(const v16h*)vp1;
    acc1 = __builtin_amdgcn_wmma_f32_16x16x32_f16(false, pa, false, bv1, (short)0, acc1, false, false);
  }
  #pragma unroll
  for (int r = 0; r < 8; ++r) {
    const int rt = (hi ? 8 : 0) + r;
    cpart[wave][rt][lh] = acc0[r];
    cpart[wave][rt][16 + lh] = acc1[r];
  }
  __syncthreads();
  for (int idx = threadIdx.x; idx < 512; idx += 128) {
    const int rt = idx >> 5, col = idx & 31;
    const float v = cpart[0][rt][col] + cpart[1][rt][col] + cpart[2][rt][col] + cpart[3][rt][col];
    const int q = q0 + rt;
    if (q < Q_) ctx16[((long)b * Q_ + q) * C_ + h * DH_ + col] = (_Float16)v;
  }
}

// ---------------------------------------------------------------------------
// LayerNorm (one block per row of 256). Optional residual; resMod=Q_ means
// residual indexed by (row % Q_) (query_feat broadcast).
// ---------------------------------------------------------------------------
__global__ __launch_bounds__(256)
void ln_kernel(const float* __restrict__ x, const float* __restrict__ res, int resMod,
               const float* __restrict__ gamma, const float* __restrict__ beta,
               float* __restrict__ y32, _Float16* __restrict__ y16) {
  __shared__ float s1[256], s2[256];
  const int row = blockIdx.x, c = threadIdx.x;
  float v = x[(long)row * C_ + c];
  if (res) v += resMod ? res[(row % resMod) * C_ + c] : res[(long)row * C_ + c];
  s1[c] = v; s2[c] = v * v;
  __syncthreads();
  for (int st = 128; st > 0; st >>= 1) {
    if (c < st) { s1[c] += s1[c + st]; s2[c] += s2[c + st]; }
    __syncthreads();
  }
  const float mu = s1[0] / (float)C_;
  const float var = s2[0] / (float)C_ - mu * mu;
  const float rstd = rsqrtf(var + 1e-5f);
  const float o = (v - mu) * rstd * gamma[c] + beta[c];
  if (y32) y32[(long)row * C_ + c] = o;
  if (y16) y16[(long)row * C_ + c] = (_Float16)o;
}

// out[b,t,q,c] = final[bt, mi[bt,q], c]
__global__ void gather_kernel(const float* __restrict__ fin, const int* __restrict__ mi,
                              float* __restrict__ out) {
  long idx = (long)blockIdx.x * blockDim.x + threadIdx.x;
  if (idx >= (long)MR_ * C_) return;
  int c = (int)(idx % C_);
  long row = idx / C_;
  int q = (int)(row % Q_);
  int bt = (int)(row / Q_);
  int m = mi[bt * Q_ + q];
  out[idx] = fin[((long)bt * Q_ + m) * C_ + c];
}

// Contrastive loss: one block per (b,t,i<NI); logsumexp over 99 negatives + pad 0.
__global__ __launch_bounds__(128)
void loss_kernel(const float* __restrict__ out, float* __restrict__ loss) {
  __shared__ float dots[128];
  const int blk = blockIdx.x;
  const int i = blk % NI_;
  const int bt = blk / NI_;
  const int t = bt % T_;
  const int b = bt / T_;
  const float* fb = out + (long)b * (T_ * Q_) * C_;
  const float* rp = fb + (long)(t * Q_ + i) * C_;
  const int ot = 1 - t;
  float d = 0.f;
  if (threadIdx.x < Q_) {
    const float* cp = fb + (long)(ot * Q_ + threadIdx.x) * C_;
    for (int c = 0; c < C_; ++c) d += rp[c] * cp[c];
  }
  dots[threadIdx.x] = d;
  __syncthreads();
  if (threadIdx.x == 0) {
    const float pv = dots[i];
    float m = 0.0f;  // pad term contributes exp(0)
    for (int j = 0; j < Q_; ++j) if (j != i) m = fmaxf(m, dots[j] - pv);
    float s = __expf(-m);
    for (int j = 0; j < Q_; ++j) if (j != i) s += __expf(dots[j] - pv - m);
    const float lse = m + __logf(s);
    atomicAdd(loss, lse / (float)(B_ * NI_ * T_));
  }
}

// ---------------------------------------------------------------------------
// Host launcher
// ---------------------------------------------------------------------------
extern "C" void kernel_launch(void* const* d_in, const int* in_sizes, int n_in,
                              void* d_out, int out_size, void* d_ws, size_t ws_size,
                              hipStream_t stream) {
  (void)in_sizes; (void)n_in; (void)out_size; (void)ws_size;

  const float* features  = (const float*)d_in[0];
  const float* tmasks    = (const float*)d_in[1];
  const float* snoise    = (const float*)d_in[2];
  const int*   mindices  = (const int*)d_in[3];
  const float* conv_w    = (const float*)d_in[4];
  const float* conv_b    = (const float*)d_in[5];
  const float* gn_gamma  = (const float*)d_in[6];
  const float* gn_beta   = (const float*)d_in[7];
  const float* qfeat     = (const float*)d_in[8];
  const float* qembed    = (const float*)d_in[9];
  const float* wq        = (const float*)d_in[10];
  const float* bq        = (const float*)d_in[11];
  const float* wk        = (const float*)d_in[12];
  const float* bk        = (const float*)d_in[13];
  const float* wv        = (const float*)d_in[14];
  const float* bv        = (const float*)d_in[15];
  const float* wo        = (const float*)d_in[16];
  const float* bo        = (const float*)d_in[17];
  const float* ca_gamma  = (const float*)d_in[18];
  const float* ca_beta   = (const float*)d_in[19];
  const float* w1        = (const float*)d_in[20];
  const float* b1        = (const float*)d_in[21];
  const float* w2        = (const float*)d_in[22];
  const float* b2        = (const float*)d_in[23];
  const float* ffn_gamma = (const float*)d_in[24];
  const float* ffn_beta  = (const float*)d_in[25];
  const float* dec_gamma = (const float*)d_in[26];
  const float* dec_beta  = (const float*)d_in[27];
  const float* embed_w   = (const float*)d_in[28];
  const float* embed_b   = (const float*)d_in[29];

  float* out = (float*)d_out;

  // ---- workspace carve-out (aligned 256B) ----
  char* w = (char*)d_ws;
  size_t off = 0;
  auto alloc = [&](size_t bytes) -> void* {
    void* p = w + off;
    off = (off + bytes + 255) & ~(size_t)255;
    return p;
  };
  const size_t bigHalves = (size_t)BT_ * HW_ * C_;  // 16.7M elements

  float*    pos      = (float*)alloc((size_t)HW_ * C_ * 4);
  _Float16* wsA      = (_Float16*)alloc(bigHalves * 2);  // featT16 -> mem16
  _Float16* wsB      = (_Float16*)alloc(bigHalves * 2);  // fpre16  -> kh16
  _Float16* wsC      = (_Float16*)alloc(bigHalves * 2);  // kin16   -> vh16T
  _Float16* cwh      = (_Float16*)alloc((size_t)C_ * C_ * 2);
  _Float16* wkh      = (_Float16*)alloc((size_t)C_ * C_ * 2);
  _Float16* wvh      = (_Float16*)alloc((size_t)C_ * C_ * 2);
  _Float16* woh      = (_Float16*)alloc((size_t)C_ * C_ * 2);
  _Float16* ewh      = (_Float16*)alloc((size_t)C_ * C_ * 2);
  _Float16* w1h      = (_Float16*)alloc((size_t)DF_ * C_ * 2);
  _Float16* w2h      = (_Float16*)alloc((size_t)DF_ * C_ * 2);
  float*    gscale   = (float*)alloc((size_t)BT_ * C_ * 4);
  float*    gshift   = (float*)alloc((size_t)BT_ * C_ * 4);
  float*    allm     = (float*)alloc((size_t)BT_ * HW_ * 4);
  unsigned char* mbuf = (unsigned char*)alloc((size_t)BT_ * Q_ * HW_);
  unsigned char* rfull = (unsigned char*)alloc((size_t)BT_ * Q_);
  _Float16* qh16     = (_Float16*)alloc((size_t)Q_ * C_ * 2);
  _Float16* ctx16    = (_Float16*)alloc((size_t)MR_ * C_ * 2);
  float*    t1       = (float*)alloc((size_t)MR_ * C_ * 4);
  float*    out1_32  = (float*)alloc((size_t)MR_ * C_ * 4);
  _Float16* out1_16  = (_Float16*)alloc((size_t)MR_ * C_ * 2);
  _Float16* h1_16    = (_Float16*)alloc((size_t)MR_ * DF_ * 2);
  float*    t2       = (float*)alloc((size_t)MR_ * C_ * 4);
  float*    out2_32  = (float*)alloc((size_t)MR_ * C_ * 4);
  _Float16* out3_16  = (_Float16*)alloc((size_t)MR_ * C_ * 2);
  float*    final32  = (float*)alloc((size_t)MR_ * C_ * 4);

  // 1. positional embedding
  pos_embed_kernel<<<(HW_ * C_ + 255) / 256, 256, 0, stream>>>(pos);

  // 2. weight converts to f16
  cvt_f16_kernel<<<(C_ * C_ + 255) / 256, 256, 0, stream>>>(conv_w, cwh, C_ * C_);
  cvt_f16_kernel<<<(C_ * C_ + 255) / 256, 256, 0, stream>>>(wk, wkh, C_ * C_);
  cvt_f16_kernel<<<(C_ * C_ + 255) / 256, 256, 0, stream>>>(wv, wvh, C_ * C_);
  cvt_f16_kernel<<<(C_ * C_ + 255) / 256, 256, 0, stream>>>(wo, woh, C_ * C_);
  cvt_f16_kernel<<<(C_ * C_ + 255) / 256, 256, 0, stream>>>(embed_w, ewh, C_ * C_);
  cvt_f16_kernel<<<(DF_ * C_ + 255) / 256, 256, 0, stream>>>(w1, w1h, DF_ * C_);
  cvt_f16_kernel<<<(DF_ * C_ + 255) / 256, 256, 0, stream>>>(w2, w2h, DF_ * C_);

  // 3. transpose features to [b][hw][c] f16 (wsA)
  transpose_f16_kernel<<<dim3(HW_ / 32, C_ / 32, BT_), 256, 0, stream>>>(features, wsA);

  // 4. conv 1x1 GEMM -> fpre16 (wsB)
  gemm_wmma_kernel<<<dim3(HW_ / 64, C_ / 64, BT_), 128, 0, stream>>>(
      wsA, cwh, conv_b, nullptr, wsB, HW_, C_, C_,
      (long)HW_ * C_, (long)HW_ * C_, 0, 0);

  // 5. groupnorm stats -> scale/shift; 6. build mem16 (wsA) & kin16 (wsC)
  gn_stats_kernel<<<BT_ * G_, 256, 0, stream>>>(wsB, gn_gamma, gn_beta, gscale, gshift);
  mem_build_kernel<<<(int)((bigHalves + 255) / 256), 256, 0, stream>>>(
      wsB, gscale, gshift, pos, wsA, wsC);

  // 7. K projection: kin16 -> kh16 (wsB, overwrites fpre16)
  gemm_wmma_kernel<<<dim3(HW_ / 64, C_ / 64, BT_), 128, 0, stream>>>(
      wsC, wkh, bk, nullptr, wsB, HW_, C_, C_,
      (long)HW_ * C_, (long)HW_ * C_, 0, 0);

  // 8. V projection: mem16 -> vh16T (wsC, transposed [c][hw], overwrites kin16)
  gemm_wmma_kernel<<<dim3(HW_ / 64, C_ / 64, BT_), 128, 0, stream>>>(
      wsA, wvh, bv, nullptr, wsC, HW_, C_, C_,
      (long)HW_ * C_, (long)HW_ * C_, 0, 1);

  // 9-11. mask pipeline
  allmask_kernel<<<(BT_ * HW_ + 255) / 256, 256, 0, stream>>>(tmasks, allm);
  maskb_kernel<<<(int)(((long)BT_ * Q_ * HW_ + 255) / 256), 256, 0, stream>>>(
      tmasks, snoise, mindices, allm, mbuf);
  rowfull_kernel<<<BT_ * Q_, 256, 0, stream>>>(mbuf, rfull);

  // 12. query projection
  qh_kernel<<<Q_, 256, 0, stream>>>(qfeat, qembed, wq, bq, qh16);

  // 13. fused masked attention
  const int smem = 16 * HW_ * (int)sizeof(float);  // 256 KB (WGP has 320 KB)
  (void)hipFuncSetAttribute(reinterpret_cast<const void*>(attention_kernel),
                            hipFuncAttributeMaxDynamicSharedMemorySize, smem);
  attention_kernel<<<dim3((Q_ + 15) / 16, NH_, BT_), 128, smem, stream>>>(
      qh16, wsB, wsC, mbuf, rfull, ctx16);

  // 14. out projection + 15. LN(query + ctx@wo^T+bo)
  gemm_wmma_kernel<<<dim3(MR_ / 64, C_ / 64, 1), 128, 0, stream>>>(
      ctx16, woh, bo, t1, nullptr, MR_, C_, C_, 0, 0, 0, 0);
  ln_kernel<<<MR_, 256, 0, stream>>>(t1, qfeat, Q_, ca_gamma, ca_beta, out1_32, out1_16);

  // 16-17. FFN
  gemm_wmma_kernel<<<dim3(MR_ / 64, DF_ / 64, 1), 128, 0, stream>>>(
      out1_16, w1h, b1, nullptr, h1_16, MR_, DF_, C_, 0, 0, 1, 0);
  gemm_wmma_kernel<<<dim3(MR_ / 64, C_ / 64, 1), 128, 0, stream>>>(
      h1_16, w2h, b2, t2, nullptr, MR_, C_, DF_, 0, 0, 0, 0);

  // 18-19. LN(ffn residual), then decoder LN
  ln_kernel<<<MR_, 256, 0, stream>>>(t2, out1_32, 0, ffn_gamma, ffn_beta, out2_32, nullptr);
  ln_kernel<<<MR_, 256, 0, stream>>>(out2_32, nullptr, 0, dec_gamma, dec_beta, nullptr, out3_16);

  // 20. embed head
  gemm_wmma_kernel<<<dim3(MR_ / 64, C_ / 64, 1), 128, 0, stream>>>(
      out3_16, ewh, embed_b, final32, nullptr, MR_, C_, C_, 0, 0, 0, 0);

  // 21. gather by mask_indices into d_out
  gather_kernel<<<(int)(((long)MR_ * C_ + 255) / 256), 256, 0, stream>>>(final32, mindices, out);

  // 22. contrastive loss into d_out[B*T*Q*C]
  (void)hipMemsetAsync(out + (size_t)MR_ * C_, 0, sizeof(float), stream);
  loss_kernel<<<B_ * T_ * NI_, 128, 0, stream>>>(out, out + (size_t)MR_ * C_);
}
